// Res_GCN_23390391894935
// MI455X (gfx1250) — compile-verified, matrix operand order checked by reading, not run on hardware
//
#include <hip/hip_runtime.h>
#include <hip/hip_bf16.h>
#include <math.h>

// Problem constants (match reference)
#define BB 2
#define DD 256
#define NN 8192
#define CC 128
#define MM (NN / 2)          // 4096 keys after maxpool2
#define BN_EPS 1e-5f

typedef __bf16 bf16t;
typedef __attribute__((ext_vector_type(16))) __bf16 v16bf;
typedef __attribute__((ext_vector_type(8)))  __bf16 v8bf;
typedef __attribute__((ext_vector_type(8)))  float  v8f;

#ifndef __has_builtin
#define __has_builtin(x) 0
#endif
#if __has_builtin(__builtin_amdgcn_tensor_load_to_lds) && \
    __has_builtin(__builtin_amdgcn_s_wait_tensorcnt)
#define HAVE_TDM 1
#endif

// float -> bf16 (round-to-nearest-even) without relying on fp casts
__device__ __forceinline__ bf16t to_bf16(float x) {
    union { float f; unsigned u; } v; v.f = x;
    unsigned r = v.u + 0x7FFFu + ((v.u >> 16) & 1u);
    union { unsigned short s; bf16t b; } o;
    o.s = (unsigned short)(r >> 16);
    return o.b;
}

__device__ __forceinline__ v8f zero8() {
    v8f z;
    for (int i = 0; i < 8; ++i) z[i] = 0.0f;
    return z;
}

// 16-bit WMMA operand fragment load (A or B): the 16 elements per lane are two
// contiguous 8-element K-runs: [half*8 .. +7] and [half*8+16 .. +23], where
// half = lane>>4 (per the CDNA5 16-bit 16x32 layout). rowPtr points at the
// start of this lane's row/column within the current 32-wide K chunk.
__device__ __forceinline__ v16bf frag_ld(const bf16t* rowPtr, int half) {
    union { v16bf v; v8bf h[2]; } u;
    u.h[0] = *(const v8bf*)(rowPtr + half * 8);
    u.h[1] = *(const v8bf*)(rowPtr + half * 8 + 16);
    return u.v;
}

__device__ __forceinline__ v8f wmma_bf16(v16bf a, v16bf b, v8f c) {
    return __builtin_amdgcn_wmma_f32_16x16x32_bf16(false, a, false, b,
                                                   (short)0, c, false, false);
}

#ifdef HAVE_TDM
typedef unsigned int u32x4 __attribute__((ext_vector_type(4)));
typedef int          i32x8 __attribute__((ext_vector_type(8)));
typedef int          i32x4 __attribute__((ext_vector_type(4)));

// LDS byte offset of a generic pointer derived from extern __shared__
__device__ __forceinline__ unsigned lds_addr_of(void* p) {
    return (unsigned)(unsigned long long)
        (__attribute__((address_space(3))) char*)p;
}

// Issue one TDM 2-D tile load: tileH rows of tileW bf16 elements from a source
// with srcStride elements between rows, into LDS with hardware row padding
// (padIntervalCode: DWORDs-per-row code (2<<c), padAmountCode: (c+1) DWORDs).
// D# packing per CDNA5 ISA ch.8 (group0 128b, group1 256b; 2-D -> groups 2/3
// zero). This toolchain's builtin takes 6 args (4 groups + spare i32x8 + cpol).
__device__ __forceinline__ void tdm_load_2d(unsigned ldsByteOff, const void* gsrc,
                                            unsigned tileW, unsigned tileH,
                                            unsigned srcStride,
                                            unsigned padIntervalCode,
                                            unsigned padAmountCode)
{
    unsigned long long ga = (unsigned long long)(size_t)gsrc;
    u32x4 g0;
    g0[0] = 1u;                                   // count=1 (valid descriptor)
    g0[1] = ldsByteOff;                           // lds_addr (bytes)
    g0[2] = (unsigned)ga;                         // global_addr[31:0] at bit 64
    g0[3] = (unsigned)((ga >> 32) & 0x01FFFFFFu)  // global_addr[56:32]
          | (2u << 30);                           // type = 2 ("image")
    i32x8 g1;
    g1[0] = (int)((1u << 16)                      // data_size = 1 -> 2 bytes
          | (1u << 20)                            // pad_enable
          | (padIntervalCode << 22)
          | (padAmountCode << 25));               // wg_mask = 0 (not in cluster)
    g1[1] = (int)((tileW & 0xFFFFu) << 16);       // tensor_dim0[15:0] @ bits 63:48
    g1[2] = (int)((tileW >> 16) | ((tileH & 0xFFFFu) << 16));  // dim0 hi | dim1 lo
    g1[3] = (int)((tileH >> 16) | (tileW << 16)); // dim1 hi | tile_dim0
    g1[4] = (int)(tileH & 0xFFFFu);               // tile_dim1 (tile_dim2 = 0)
    g1[5] = (int)srcStride;                       // tensor_dim0_stride[31:0]
    g1[6] = 0;                                    // stride hi | dim1_stride lo
    g1[7] = 0;
    i32x4 gz; gz[0] = 0; gz[1] = 0; gz[2] = 0; gz[3] = 0;
    i32x8 gz8;
    for (int i = 0; i < 8; ++i) gz8[i] = 0;
    __builtin_amdgcn_tensor_load_to_lds(g0, g1, gz, gz, gz8, 0);
}
#endif // HAVE_TDM

// ---------------------------------------------------------------------------
// Kernel 1: 1x1-conv projections (theta / phi / g) with fused bias + maxpool2.
// grid = (N/64, B, 3). mode 0: theta -> thetaT (B,N,CC)
//                      mode 1: phi   -> phiT   (B,MM,CC)   (pooled)
//                      mode 2: g     -> gC     (B,CC,MM)   (pooled, c-major)
// ---------------------------------------------------------------------------
#define K1_LDW 264   // padded K stride (256 + 8 bf16)

extern "C" __global__ __launch_bounds__(256)
void k_proj(const float* __restrict__ f,
            const float* __restrict__ thw, const float* __restrict__ thb,
            const float* __restrict__ phw, const float* __restrict__ phb,
            const float* __restrict__ gw,  const float* __restrict__ gb,
            bf16t* __restrict__ thetaT, bf16t* __restrict__ phiT,
            bf16t* __restrict__ gC)
{
    extern __shared__ char smem[];
    bf16t* wLds = (bf16t*)smem;                                 // 128 x 264 bf16
    bf16t* fLds = (bf16t*)(smem + (size_t)CC * K1_LDW * 2);     // 64 x 264 bf16
    float* oLds = (float*)(smem + (size_t)CC * K1_LDW * 2);     // overlay: 128x64 f32

    const int t    = threadIdx.x;
    const int b    = blockIdx.y;
    const int mode = blockIdx.z;
    const int n0   = blockIdx.x * 64;

    const float* w    = (mode == 0) ? thw : (mode == 1) ? phw : gw;
    const float* bias = (mode == 0) ? thb : (mode == 1) ? phb : gb;

    // stage weights (CC x DD) -> bf16 (f32 source: convert, cannot DMA)
    for (int idx = t; idx < CC * DD; idx += 256) {
        int c = idx >> 8, d = idx & 255;
        wLds[c * K1_LDW + d] = to_bf16(w[idx]);
    }
    // stage f tile transposed: fLds[j][d] = f[b, d, n0+j]
    for (int idx = t; idx < DD * 64; idx += 256) {
        int d = idx >> 6, j = idx & 63;
        fLds[j * K1_LDW + d] = to_bf16(f[((size_t)b * DD + d) * NN + n0 + j]);
    }
    __syncthreads();

    const int wave = t >> 5, lane = t & 31, lm = lane & 15, half = lane >> 4;
    v8f acc[4];
    for (int s = 0; s < 4; ++s) acc[s] = zero8();

    // out[16ci x 64n] per wave, K = 256 in 8 chunks of 32
    for (int kk = 0; kk < DD / 32; ++kk) {
        v16bf a = frag_ld(wLds + (wave * 16 + lm) * K1_LDW + kk * 32, half);
#pragma unroll
        for (int s = 0; s < 4; ++s) {
            v16bf bm = frag_ld(fLds + (s * 16 + lm) * K1_LDW + kk * 32, half);
            acc[s] = wmma_bf16(a, bm, acc[s]);
        }
    }

    __syncthreads();   // everyone done reading fLds before overlaying oLds
#pragma unroll
    for (int s = 0; s < 4; ++s)
#pragma unroll
        for (int r = 0; r < 8; ++r) {
            int ci = wave * 16 + r + 8 * half;          // C/D layout row
            oLds[ci * 64 + s * 16 + lm] = acc[s][r] + bias[ci];
        }
    __syncthreads();

    if (mode == 0) {                 // theta: full resolution, (N, CC)
        for (int idx = t; idx < CC * 64; idx += 256) {
            int c = idx & 127, j = idx >> 7;
            thetaT[((size_t)b * NN + n0 + j) * CC + c] = to_bf16(oLds[c * 64 + j]);
        }
    } else if (mode == 1) {          // phi: maxpool2 -> (MM, CC)
        for (int idx = t; idx < CC * 32; idx += 256) {
            int c = idx & 127, jj = idx >> 7;
            float v = fmaxf(oLds[c * 64 + 2 * jj], oLds[c * 64 + 2 * jj + 1]);
            phiT[((size_t)b * MM + (n0 >> 1) + jj) * CC + c] = to_bf16(v);
        }
    } else {                         // g: maxpool2 -> (CC, MM) c-major
        for (int idx = t; idx < CC * 32; idx += 256) {
            int jj = idx & 31, c = idx >> 5;
            float v = fmaxf(oLds[c * 64 + 2 * jj], oLds[c * 64 + 2 * jj + 1]);
            gC[((size_t)b * CC + c) * MM + (n0 >> 1) + jj] = to_bf16(v);
        }
    }
}

// ---------------------------------------------------------------------------
// Kernel 2: fused flash attention.  grid = (N/128, B), 256 threads (8 waves).
// Each wave owns 16 query rows; loop over 64-key chunks with online softmax.
// Tile staging via TDM (tensor_load_to_lds) when available.
// ---------------------------------------------------------------------------
#define K2_LDQ 136   // stride for [* x 128c] tiles (256B row + 16B pad)
#define K2_LDM 72    // stride for [* x 64m] tiles (128B row + 16B pad)
#define K2_OFF_PH  ((size_t)128 * K2_LDQ * 2)                       // 34816
#define K2_OFF_G   (K2_OFF_PH + (size_t)64 * K2_LDQ * 2)            // +17408
#define K2_OFF_P   (K2_OFF_G  + (size_t)128 * K2_LDM * 2)           // +18432
#define K2_LDS_TOT (K2_OFF_P  + (size_t)8 * 16 * K2_LDM * 2)        // +18432

extern "C" __global__ __launch_bounds__(256)
void k_attn(const bf16t* __restrict__ thetaT, const bf16t* __restrict__ phiT,
            const bf16t* __restrict__ gC, bf16t* __restrict__ yT)
{
    extern __shared__ char smem[];
    bf16t* qLds  = (bf16t*)smem;                    // 128 q x 128 c
    bf16t* phLds = (bf16t*)(smem + K2_OFF_PH);      //  64 m x 128 c
    bf16t* gLds  = (bf16t*)(smem + K2_OFF_G);       // 128 c x  64 m
    bf16t* pLds  = (bf16t*)(smem + K2_OFF_P);       // per-wave 16 q x 64 m

    const int t = threadIdx.x, wave = t >> 5, lane = t & 31;
    const int lm = lane & 15, half = lane >> 4;
    const int b  = blockIdx.y;
    const int q0 = blockIdx.x * 128;

    // stage query tile (already bf16)
#ifdef HAVE_TDM
    if (wave == 0) {
        // 128 rows x 128 bf16, row pad: 64 DWORDs (code 5) + 4 DWORDs (code 3)
        tdm_load_2d(lds_addr_of(qLds), thetaT + ((size_t)b * NN + q0) * CC,
                    128u, 128u, (unsigned)CC, 5u, 3u);
    }
#else
    {
        const uint4* src = (const uint4*)(thetaT + ((size_t)b * NN + q0) * CC);
        for (int idx = t; idx < 128 * 16; idx += 256) {
            int q = idx >> 4, c8 = idx & 15;
            *(uint4*)(qLds + q * K2_LDQ + c8 * 8) = src[idx];
        }
    }
#endif

    v8f   O[8];
    float Mx[8], L[8];
    for (int i = 0; i < 8; ++i) { O[i] = zero8(); Mx[i] = -3.0e38f; L[i] = 0.0f; }

    const int   qw = wave * 16;
    bf16t*      pW = pLds + (size_t)wave * 16 * K2_LDM;

    for (int kt = 0; kt < MM / 64; ++kt) {
        __syncthreads();   // previous chunk's K/V reads complete
#ifdef HAVE_TDM
        if (wave == 0) {
            // phi tile: 64 rows x 128 bf16 (pad codes 5,3)
            tdm_load_2d(lds_addr_of(phLds),
                        phiT + ((size_t)b * MM + kt * 64) * CC,
                        128u, 64u, (unsigned)CC, 5u, 3u);
            // g tile: 128 rows x 64 bf16, src row stride MM (pad: 32 DW -> code 4)
            tdm_load_2d(lds_addr_of(gLds),
                        gC + ((size_t)b * CC) * MM + kt * 64,
                        64u, 128u, (unsigned)MM, 4u, 3u);
            __builtin_amdgcn_s_wait_tensorcnt(0);
        }
#else
        {   // stage phi tile [64 m x 128 c]
            const uint4* src = (const uint4*)(phiT + ((size_t)b * MM + kt * 64) * CC);
            for (int idx = t; idx < 64 * 16; idx += 256) {
                int m = idx >> 4, c8 = idx & 15;
                *(uint4*)(phLds + m * K2_LDQ + c8 * 8) = src[idx];
            }
            // stage g tile [128 c x 64 m]
            for (int idx = t; idx < 128 * 8; idx += 256) {
                int c = idx >> 3, m8 = idx & 7;
                *(uint4*)(gLds + c * K2_LDM + m8 * 8) =
                    *(const uint4*)(gC + ((size_t)b * CC + c) * MM + kt * 64 + m8 * 8);
            }
            if (kt + 1 < MM / 64) {   // prefetch next chunk into GL2/L0
                __builtin_prefetch(phiT + ((size_t)b * MM + (kt + 1) * 64) * CC + t * 32, 0, 0);
                __builtin_prefetch(gC + ((size_t)b * CC + (t >> 1)) * MM + (kt + 1) * 64, 0, 0);
            }
        }
#endif
        __syncthreads();

        // S(16q x 64m) = Q x K^T, K-dim = CC = 128
        v8f S[4];
        for (int s = 0; s < 4; ++s) S[s] = zero8();
#pragma unroll
        for (int kk = 0; kk < 4; ++kk) {
            v16bf a = frag_ld(qLds + (qw + lm) * K2_LDQ + kk * 32, half);
#pragma unroll
            for (int s = 0; s < 4; ++s) {
                v16bf bm = frag_ld(phLds + (s * 16 + lm) * K2_LDQ + kk * 32, half);
                S[s] = wmma_bf16(a, bm, S[s]);
            }
        }

        // online softmax per accumulator row r (actual row = r + 8*half)
#pragma unroll
        for (int r = 0; r < 8; ++r) {
            float rm = fmaxf(fmaxf(S[0][r], S[1][r]), fmaxf(S[2][r], S[3][r]));
            rm = fmaxf(rm, __shfl_xor(rm, 1, 32));
            rm = fmaxf(rm, __shfl_xor(rm, 2, 32));
            rm = fmaxf(rm, __shfl_xor(rm, 4, 32));
            rm = fmaxf(rm, __shfl_xor(rm, 8, 32));
            float Mn   = fmaxf(Mx[r], rm);
            float corr = __expf(Mx[r] - Mn);
            float ls = 0.0f;
#pragma unroll
            for (int s = 0; s < 4; ++s) {
                float p = __expf(S[s][r] - Mn);
                ls += p;
                pW[(r + 8 * half) * K2_LDM + s * 16 + lm] = to_bf16(p);
            }
            ls += __shfl_xor(ls, 1, 32);
            ls += __shfl_xor(ls, 2, 32);
            ls += __shfl_xor(ls, 4, 32);
            ls += __shfl_xor(ls, 8, 32);
            L[r]  = L[r] * corr + ls;
            Mx[r] = Mn;
#pragma unroll
            for (int cs = 0; cs < 8; ++cs) O[cs][r] *= corr;
        }
        // pW writes (this wave) must land before the A-fragment reads below
        asm volatile("s_wait_dscnt 0x0" ::: "memory");

        // O(16q x 128c) += P(16q x 64m) x g^T(64m x 128c)
#pragma unroll
        for (int kk2 = 0; kk2 < 2; ++kk2) {
            v16bf aP = frag_ld(pW + lm * K2_LDM + kk2 * 32, half);
#pragma unroll
            for (int cs = 0; cs < 8; ++cs) {
                v16bf bG = frag_ld(gLds + (cs * 16 + lm) * K2_LDM + kk2 * 32, half);
                O[cs] = wmma_bf16(aP, bG, O[cs]);
            }
        }
    }

    // normalize and store y transposed: yT[b, q, c]
#pragma unroll
    for (int r = 0; r < 8; ++r) {
        float inv = 1.0f / L[r];
        size_t row = (size_t)b * NN + q0 + qw + r + 8 * half;
#pragma unroll
        for (int cs = 0; cs < 8; ++cs)
            yT[row * CC + cs * 16 + lm] = to_bf16(O[cs][r] * inv);
    }
}

// ---------------------------------------------------------------------------
// Kernel 3a: Wy = W_w @ y + W_b (f32 out) + per-channel BN partial sums.
// grid = (N/64, B), 256 threads; each wave does 32 d-rows x 64 n-cols.
// ---------------------------------------------------------------------------
#define K3_LDC 136

extern "C" __global__ __launch_bounds__(256)
void k_wconv(const bf16t* __restrict__ yT, const float* __restrict__ Ww,
             const float* __restrict__ Wb, float* __restrict__ Wy,
             float* __restrict__ bnSum, float* __restrict__ bnSq)
{
    extern __shared__ char smem[];
    bf16t* wLds = (bf16t*)smem;                                  // 256 x 136 bf16
    bf16t* yLds = (bf16t*)(smem + (size_t)DD * K3_LDC * 2);      //  64 x 136 bf16

    const int t = threadIdx.x, wave = t >> 5, lane = t & 31;
    const int lm = lane & 15, half = lane >> 4;
    const int b = blockIdx.y, n0 = blockIdx.x * 64;

#ifdef HAVE_TDM
    if (wave == 0) {   // y tile: 64 rows x 128 bf16 (pad codes 5,3)
        tdm_load_2d(lds_addr_of(yLds), yT + ((size_t)b * NN + n0) * CC,
                    128u, 64u, (unsigned)CC, 5u, 3u);
    }
#else
    {
        const uint4* src = (const uint4*)(yT + ((size_t)b * NN + n0) * CC);
        for (int idx = t; idx < 64 * 16; idx += 256) {
            int j = idx >> 4, c8 = idx & 15;
            *(uint4*)(yLds + j * K3_LDC + c8 * 8) = src[idx];
        }
    }
#endif
    for (int idx = t; idx < DD * CC; idx += 256) {
        int d = idx >> 7, c = idx & 127;
        wLds[d * K3_LDC + c] = to_bf16(Ww[idx]);
    }
#ifdef HAVE_TDM
    if (wave == 0) __builtin_amdgcn_s_wait_tensorcnt(0);
#endif
    __syncthreads();

    v8f acc[2][4];
    for (int i = 0; i < 2; ++i)
        for (int s = 0; s < 4; ++s) acc[i][s] = zero8();

#pragma unroll
    for (int kk = 0; kk < 4; ++kk) {
        v16bf a0 = frag_ld(wLds + (wave * 32 + lm) * K3_LDC + kk * 32, half);
        v16bf a1 = frag_ld(wLds + (wave * 32 + 16 + lm) * K3_LDC + kk * 32, half);
#pragma unroll
        for (int s = 0; s < 4; ++s) {
            v16bf bm = frag_ld(yLds + (s * 16 + lm) * K3_LDC + kk * 32, half);
            acc[0][s] = wmma_bf16(a0, bm, acc[0][s]);
            acc[1][s] = wmma_bf16(a1, bm, acc[1][s]);
        }
    }

#pragma unroll
    for (int rt = 0; rt < 2; ++rt) {
#pragma unroll
        for (int r = 0; r < 8; ++r) {
            int d = wave * 32 + rt * 16 + r + 8 * half;
            float bias = Wb[d];
            float ps = 0.0f, ps2 = 0.0f;
#pragma unroll
            for (int s = 0; s < 4; ++s) {
                float v = acc[rt][s][r] + bias;
                Wy[((size_t)b * DD + d) * NN + n0 + s * 16 + lm] = v;
                ps += v; ps2 += v * v;
            }
            ps  += __shfl_xor(ps, 1, 32);  ps2 += __shfl_xor(ps2, 1, 32);
            ps  += __shfl_xor(ps, 2, 32);  ps2 += __shfl_xor(ps2, 2, 32);
            ps  += __shfl_xor(ps, 4, 32);  ps2 += __shfl_xor(ps2, 4, 32);
            ps  += __shfl_xor(ps, 8, 32);  ps2 += __shfl_xor(ps2, 8, 32);
            if (lm == 0) {
                atomicAdd(&bnSum[d], ps);
                atomicAdd(&bnSq[d],  ps2);
            }
        }
    }
}

// ---------------------------------------------------------------------------
// Kernel 3b: BN normalize (training-mode batch stats) + residual add.
// ---------------------------------------------------------------------------
extern "C" __global__ __launch_bounds__(256)
void k_bnres(const float* __restrict__ Wy, const float* __restrict__ f,
             const float* __restrict__ bnSum, const float* __restrict__ bnSq,
             const float* __restrict__ gamma, const float* __restrict__ beta,
             float* __restrict__ out)
{
    const float invN = 1.0f / (float)(BB * NN);
    size_t i4 = (size_t)blockIdx.x * blockDim.x + threadIdx.x;
    size_t total4 = (size_t)BB * DD * NN / 4;
    if (i4 >= total4) return;
    size_t i = i4 * 4;
    int d = (int)((i / NN) % DD);

    float mean = bnSum[d] * invN;
    float var  = fmaxf(bnSq[d] * invN - mean * mean, 0.0f);
    float sc   = rsqrtf(var + BN_EPS) * gamma[d];
    float sh   = beta[d] - mean * sc;

    float4 wy = *(const float4*)(Wy + i);
    float4 ff = *(const float4*)(f + i);
    float4 o;
    o.x = wy.x * sc + sh + ff.x;
    o.y = wy.y * sc + sh + ff.y;
    o.z = wy.z * sc + sh + ff.z;
    o.w = wy.w * sc + sh + ff.w;
    *(float4*)(out + i) = o;
}

// ---------------------------------------------------------------------------
extern "C" void kernel_launch(void* const* d_in, const int* in_sizes, int n_in,
                              void* d_out, int out_size, void* d_ws, size_t ws_size,
                              hipStream_t stream)
{
    const float* f     = (const float*)d_in[0];
    const float* g_w   = (const float*)d_in[1];
    const float* g_b   = (const float*)d_in[2];
    const float* th_w  = (const float*)d_in[3];
    const float* th_b  = (const float*)d_in[4];
    const float* ph_w  = (const float*)d_in[5];
    const float* ph_b  = (const float*)d_in[6];
    const float* W_w   = (const float*)d_in[7];
    const float* W_b   = (const float*)d_in[8];
    const float* gamma = (const float*)d_in[9];
    const float* beta  = (const float*)d_in[10];
    float* out = (float*)d_out;

    char* ws = (char*)d_ws;
    size_t off = 0;
    auto take = [&](size_t bytes) -> char* {
        char* p = ws + off;
        off = (off + bytes + 255) & ~(size_t)255;
        return p;
    };
    bf16t* thetaT = (bf16t*)take((size_t)BB * NN * CC * 2);   // (B, N, CC)
    bf16t* phiT   = (bf16t*)take((size_t)BB * MM * CC * 2);   // (B, MM, CC)
    bf16t* gCm    = (bf16t*)take((size_t)BB * CC * MM * 2);   // (B, CC, MM)
    bf16t* yT     = (bf16t*)take((size_t)BB * NN * CC * 2);   // (B, N, CC)
    float* Wy     = (float*)take((size_t)BB * DD * NN * 4);   // (B, D, N)
    float* bnSum  = (float*)take((size_t)DD * 4);
    float* bnSq   = (float*)take((size_t)DD * 4);

    (void)hipMemsetAsync(bnSum, 0, DD * 4, stream);
    (void)hipMemsetAsync(bnSq,  0, DD * 4, stream);

    size_t lds1 = (size_t)CC * K1_LDW * 2 + (size_t)64 * K1_LDW * 2;   // ~101 KB
    k_proj<<<dim3(NN / 64, BB, 3), 256, lds1, stream>>>(
        f, th_w, th_b, ph_w, ph_b, g_w, g_b, thetaT, phiT, gCm);

    k_attn<<<dim3(NN / 128, BB), 256, K2_LDS_TOT, stream>>>(thetaT, phiT, gCm, yT);

    size_t lds3 = (size_t)DD * K3_LDC * 2 + (size_t)64 * K3_LDC * 2;   // ~85 KB
    k_wconv<<<dim3(NN / 64, BB), 256, lds3, stream>>>(yT, W_w, W_b, Wy, bnSum, bnSq);

    size_t total4 = (size_t)BB * DD * NN / 4;
    k_bnres<<<dim3((unsigned)((total4 + 255) / 256)), 256, 0, stream>>>(
        Wy, f, bnSum, bnSq, gamma, beta, out);
}
